// Qwen35VisionAttention_80934363726547
// MI455X (gfx1250) — matrix operand order repaired and checked by reading, not compiled, not run on hardware
//
#include <hip/hip_runtime.h>
#include <hip/hip_bf16.h>
#include <stdint.h>

// ---------------------------------------------------------------------------
// MI455X (gfx1250) CDNA5 implementation: all matmuls through
// v_wmma_f32_16x16x32_bf16 (wave32).  Compute-bound workload (~75 GFLOP over
// ~35MB of data that fits in L2).  GEMM operand staging uses the CDNA5
// async-to-LDS path (global_load_async_to_lds_b128, ASYNCcnt-tracked).
// ---------------------------------------------------------------------------

typedef __attribute__((ext_vector_type(8)))  __bf16 v8bf;
typedef __attribute__((ext_vector_type(16))) __bf16 v16bf;
typedef __attribute__((ext_vector_type(8)))  float  v8f;

static constexpr int S_   = 4096;
static constexpr int D_   = 1280;
static constexpr int H_   = 16;
static constexpr int HD_  = 80;     // head dim
static constexpr int SEG_ = 1024;   // segment length (cu_seqlens is uniform)
static constexpr int DQKV = 3 * D_; // 3840

__device__ __forceinline__ v8bf ld8(const __bf16* p) { return *(const v8bf*)p; }

__device__ __forceinline__ v16bf cat16(v8bf lo, v8bf hi) {
  return __builtin_shufflevector(lo, hi, 0, 1, 2, 3, 4, 5, 6, 7,
                                         8, 9, 10, 11, 12, 13, 14, 15);
}

// A-matrix fragment (16x32 bf16): lane holds row=lane&15; k-values are two
// 8-element runs at k = 8*hi and k = 16 + 8*hi (ISA 16-bit A layout).
__device__ __forceinline__ v16bf load_a_frag(const __bf16* row, int hi) {
  const __bf16* p = row + 8 * hi;
  return cat16(ld8(p), ld8(p + 16));
}

// B-matrix fragment (32x16 bf16): lane holds col n=lane&15; k-values are two
// 8-element runs at k = 8*hi and k = 16 + 8*hi (matches A layout).
__device__ __forceinline__ v16bf load_b_frag(const __bf16* row, int hi) {
  const __bf16* p = row + 8 * hi;
  return cat16(ld8(p), ld8(p + 16));
}

__device__ __forceinline__ v8f wmma_bf16(v16bf a, v16bf b, v8f c) {
  return __builtin_amdgcn_wmma_f32_16x16x32_bf16(
      /*neg_a=*/false, a, /*neg_b=*/false, b,
      /*c_mod=*/(short)0, c, /*reuse_a=*/false, /*reuse_b=*/false);
}

// CDNA5 async copy: 16 bytes/lane, global -> LDS, GVS addressing
// (SGPR 64-bit base + per-lane 32-bit offset).  Tracked by ASYNCcnt.
__device__ __forceinline__ void async_g2l_b128(uint32_t lds_byte_off,
                                               uint64_t gbase,
                                               uint32_t goff_bytes) {
  asm volatile("global_load_async_to_lds_b128 %0, %1, %2"
               :
               : "v"(lds_byte_off), "v"(goff_bytes), "s"(gbase)
               : "memory");
}

__device__ __forceinline__ void wait_asynccnt0() {
  asm volatile("s_wait_asynccnt 0x0" ::: "memory");
}

// ---------------------------------------------------------------------------
// fp32 -> bf16 conversion
// ---------------------------------------------------------------------------
__global__ void cvt_bf16_kernel(const float* __restrict__ src,
                                __bf16* __restrict__ dst, int n) {
  for (int i = blockIdx.x * blockDim.x + threadIdx.x; i < n;
       i += gridDim.x * blockDim.x)
    dst[i] = (__bf16)src[i];
}

// ---------------------------------------------------------------------------
// WMMA GEMM:  C[M x N] = A[M x K] * W[N x K]^T + bias
//   A, W bf16 row-major.  Output bf16 (QKV) or fp32 (projection).
//   Block: 256 threads (8 waves), 128x128 tile.  Wave grid 2x4, each wave
//   computes 64x32 via 4x2 accumulators.  K staged in LDS 32 at a time via
//   global_load_async_to_lds_b128.
// ---------------------------------------------------------------------------
template <bool OUT_BF16>
__global__ __launch_bounds__(256) void gemm_wmma_kernel(
    const __bf16* __restrict__ A, const __bf16* __restrict__ W,
    const float* __restrict__ bias, __bf16* __restrict__ outB,
    float* __restrict__ outF, int M, int N, int K) {
  __shared__ __bf16 Asm[128 * 32];
  __shared__ __bf16 Bsm[128 * 32];

  const int tid    = threadIdx.x;
  const int lane   = tid & 31;
  const int wave   = tid >> 5;   // 0..7
  const int wm     = wave >> 2;  // 0..1  (64 rows each)
  const int wn     = wave & 3;   // 0..3  (32 cols each)
  const int lane16 = lane & 15;
  const int hi     = lane >> 4;

  const int m0 = blockIdx.y * 128;
  const int n0 = blockIdx.x * 128;

  v8f acc[4][2];
#pragma unroll
  for (int mt = 0; mt < 4; ++mt)
#pragma unroll
    for (int nt = 0; nt < 2; ++nt) acc[mt][nt] = (v8f){};

  // cooperative staging: each thread moves 16 contiguous bf16 of A and W
  const int srow = tid >> 1;         // 0..127
  const int scol = (tid & 1) * 16;   // 0 or 16

  const uint32_t ldsA = (uint32_t)(uintptr_t)&Asm[srow * 32 + scol];
  const uint32_t ldsB = (uint32_t)(uintptr_t)&Bsm[srow * 32 + scol];
  const uint64_t baseA = (uint64_t)(uintptr_t)A;
  const uint64_t baseW = (uint64_t)(uintptr_t)W;
  const uint32_t rowOffA = (uint32_t)(((size_t)(m0 + srow) * K + scol) * 2);
  const uint32_t rowOffW = (uint32_t)(((size_t)(n0 + srow) * K + scol) * 2);

  for (int kk = 0; kk < K; kk += 32) {
    __syncthreads();  // waits dscnt for prior fragment reads before overwrite
    {
      const uint32_t ga = rowOffA + (uint32_t)(kk * 2);
      const uint32_t gw = rowOffW + (uint32_t)(kk * 2);
      async_g2l_b128(ldsA,      baseA, ga);
      async_g2l_b128(ldsA + 16, baseA, ga + 16);
      async_g2l_b128(ldsB,      baseW, gw);
      async_g2l_b128(ldsB + 16, baseW, gw + 16);
      if (kk + 32 < K) {  // global_prefetch_b8 for the next K-slab
        __builtin_prefetch(A + ((size_t)(m0 + srow) * K + kk + 32 + scol), 0, 0);
        __builtin_prefetch(W + ((size_t)(n0 + srow) * K + kk + 32 + scol), 0, 0);
      }
    }
    wait_asynccnt0();
    __syncthreads();

    v16bf bfrag[2];
#pragma unroll
    for (int nt = 0; nt < 2; ++nt)
      bfrag[nt] = load_b_frag(&Bsm[(wn * 32 + nt * 16 + lane16) * 32], hi);
#pragma unroll
    for (int mt = 0; mt < 4; ++mt) {
      v16bf afrag = load_a_frag(&Asm[(wm * 64 + mt * 16 + lane16) * 32], hi);
#pragma unroll
      for (int nt = 0; nt < 2; ++nt)
        acc[mt][nt] = wmma_bf16(afrag, bfrag[nt], acc[mt][nt]);
    }
  }

  // epilogue: C row = vgpr + 8*hi, col = lane16 (ISA 32-bit C/D layout)
#pragma unroll
  for (int mt = 0; mt < 4; ++mt) {
#pragma unroll
    for (int nt = 0; nt < 2; ++nt) {
      const int col     = n0 + wn * 32 + nt * 16 + lane16;
      const float bv    = bias[col];
      const int rowbase = m0 + wm * 64 + mt * 16 + hi * 8;
#pragma unroll
      for (int r = 0; r < 8; ++r) {
        const float v    = acc[mt][nt][r] + bv;
        const size_t off = (size_t)(rowbase + r) * N + col;
        if (OUT_BF16)
          outB[off] = (__bf16)v;
        else
          outF[off] = v;
      }
    }
  }
}

// ---------------------------------------------------------------------------
// RoPE, in-place on the bf16 QKV buffer (q and k only).  Each thread owns a
// (d, d+40) pair of one (s, h), so no cross-thread races.
// ---------------------------------------------------------------------------
__global__ __launch_bounds__(256) void rope_kernel(
    __bf16* __restrict__ qkv, const float* __restrict__ cosp,
    const float* __restrict__ sinp) {
  const int idx = blockIdx.x * blockDim.x + threadIdx.x;
  if (idx >= S_ * H_ * (HD_ / 2)) return;
  const int d = idx % 40;
  const int h = (idx / 40) % H_;
  const int s = idx / (40 * H_);

  const float c0 = cosp[s * HD_ + d];
  const float s0 = sinp[s * HD_ + d];
  const float c1 = cosp[s * HD_ + d + 40];
  const float s1 = sinp[s * HD_ + d + 40];

  size_t base = (size_t)s * DQKV + h * HD_;
#pragma unroll
  for (int which = 0; which < 2; ++which) {  // q then k
    const float a = (float)qkv[base + d];
    const float b = (float)qkv[base + d + 40];
    qkv[base + d]      = (__bf16)(a * c0 - b * s0);
    qkv[base + d + 40] = (__bf16)(b * c1 + a * s1);
    base += D_;  // advance to k plane
  }
}

// ---------------------------------------------------------------------------
// Segment-local flash attention, all-WMMA.
//   grid = (q_tile 0..15, segment 0..3, head 0..15), block = 128 (4 waves).
//   Each wave owns 16 query rows; the workgroup shares LDS-staged K (32x96,
//   head-dim zero-padded 80->96) and V^T (80x32) per 32-key block.
// ---------------------------------------------------------------------------
__global__ __launch_bounds__(128) void attn_kernel(
    const __bf16* __restrict__ qkv, __bf16* __restrict__ out) {
  __shared__ __bf16 Ksm[32 * 96];      // [key][kdim padded to 96]
  __shared__ __bf16 Vsm[80 * 32];      // transposed: [vdim][key]
  __shared__ __bf16 Psm[4][16 * 32];   // per-wave P tile

  const int tid    = threadIdx.x;
  const int lane   = tid & 31;
  const int wave   = tid >> 5;  // 0..3
  const int lane16 = lane & 15;
  const int hi     = lane >> 4;

  const int seg     = blockIdx.y;
  const int h       = blockIdx.z;
  const int segbase = seg * SEG_;
  const int qrow0   = segbase + blockIdx.x * 64 + wave * 16;
  const float scale = 0.11180339887498949f;  // 1/sqrt(80)

  // Q fragments for the 3 k-chunks (head dim 80 zero-padded to 96)
  v16bf qf[3];
  {
    const __bf16* qrowp = qkv + (size_t)(qrow0 + lane16) * DQKV + h * HD_;
#pragma unroll
    for (int c = 0; c < 3; ++c) {
      v16bf v;
#pragma unroll
      for (int i = 0; i < 16; ++i) {
        const int kd = c * 32 + (i < 8 ? i : i + 8) + 8 * hi;
        v[i] = (kd < HD_) ? qrowp[kd] : (__bf16)0.0f;
      }
      qf[c] = v;
    }
  }

  float m_r[8], l_r[8];
  v8f acc_o[5];
#pragma unroll
  for (int r = 0; r < 8; ++r) { m_r[r] = -1.0e30f; l_r[r] = 0.0f; }
#pragma unroll
  for (int t = 0; t < 5; ++t) acc_o[t] = (v8f){};

  for (int kb = 0; kb < SEG_ / 32; ++kb) {
    const int key0 = segbase + kb * 32;
    __syncthreads();
    // Stage K (zero-padded) and V^T cooperatively
    for (int idx = tid; idx < 32 * 96; idx += 128) {
      const int krow = idx / 96, kd = idx % 96;
      Ksm[idx] = (kd < HD_)
          ? qkv[(size_t)(key0 + krow) * DQKV + D_ + h * HD_ + kd]
          : (__bf16)0.0f;
    }
    for (int idx = tid; idx < 80 * 32; idx += 128) {
      const int d = idx >> 5, kcol = idx & 31;
      Vsm[idx] = qkv[(size_t)(key0 + kcol) * DQKV + 2 * D_ + h * HD_ + d];
    }
    __syncthreads();

    // scores: two 16x16 tiles over 32 keys, contraction over padded head dim
    v8f s0 = (v8f){}, s1 = (v8f){};
#pragma unroll
    for (int c = 0; c < 3; ++c) {
      v16bf b0 = load_b_frag(&Ksm[lane16 * 96 + c * 32], hi);
      v16bf b1 = load_b_frag(&Ksm[(16 + lane16) * 96 + c * 32], hi);
      s0 = wmma_bf16(qf[c], b0, s0);
      s1 = wmma_bf16(qf[c], b1, s1);
    }

    // online softmax; rows r+8*hi reduce across their 16-lane half
    float alpha[8];
#pragma unroll
    for (int r = 0; r < 8; ++r) {
      const float v0 = s0[r] * scale;
      const float v1 = s1[r] * scale;
      float mx = fmaxf(v0, v1);
#pragma unroll
      for (int off = 8; off >= 1; off >>= 1)
        mx = fmaxf(mx, __shfl_xor(mx, off, 32));
      const float mnew = fmaxf(m_r[r], mx);
      alpha[r] = __expf(m_r[r] - mnew);
      m_r[r] = mnew;
      const float p0 = __expf(v0 - mnew);
      const float p1 = __expf(v1 - mnew);
      s0[r] = p0;
      s1[r] = p1;
      float sum = p0 + p1;
#pragma unroll
      for (int off = 8; off >= 1; off >>= 1)
        sum += __shfl_xor(sum, off, 32);
      l_r[r] = l_r[r] * alpha[r] + sum;
    }
#pragma unroll
    for (int t = 0; t < 5; ++t)
#pragma unroll
      for (int r = 0; r < 8; ++r) acc_o[t][r] *= alpha[r];

    // C-layout -> A-layout transpose of P through per-wave LDS tile
    __bf16* Pw = &Psm[wave][0];
#pragma unroll
    for (int r = 0; r < 8; ++r) {
      const int row = r + 8 * hi;
      Pw[row * 32 + lane16]      = (__bf16)s0[r];
      Pw[row * 32 + 16 + lane16] = (__bf16)s1[r];
    }
    asm volatile("s_wait_dscnt 0x0" ::: "memory");  // intra-wave LDS ordering

    // P fragment (A-layout, K=32 keys)
    const __bf16* pp = &Pw[lane16 * 32 + 8 * hi];
    v16bf pf = cat16(ld8(pp), ld8(pp + 16));

    // acc_o += P @ V   (5 N-tiles cover vdim 80)
#pragma unroll
    for (int t = 0; t < 5; ++t) {
      v16bf vb = load_b_frag(&Vsm[(t * 16 + lane16) * 32], hi);
      acc_o[t] = wmma_bf16(pf, vb, acc_o[t]);
    }
  }

  // normalize and store bf16
#pragma unroll
  for (int t = 0; t < 5; ++t) {
#pragma unroll
    for (int r = 0; r < 8; ++r) {
      const int row = qrow0 + r + 8 * hi;
      const int col = h * HD_ + t * 16 + lane16;
      out[(size_t)row * D_ + col] = (__bf16)(acc_o[t][r] / l_r[r]);
    }
  }
}

// ---------------------------------------------------------------------------
// Host-side launcher
// ---------------------------------------------------------------------------
extern "C" void kernel_launch(void* const* d_in, const int* in_sizes, int n_in,
                              void* d_out, int out_size, void* d_ws,
                              size_t ws_size, hipStream_t stream) {
  const float* x      = (const float*)d_in[0];
  const float* qkv_w  = (const float*)d_in[1];
  const float* qkv_b  = (const float*)d_in[2];
  const float* proj_w = (const float*)d_in[3];
  const float* proj_b = (const float*)d_in[4];
  const float* poscos = (const float*)d_in[5];
  const float* possin = (const float*)d_in[6];
  (void)in_sizes; (void)n_in; (void)out_size; (void)ws_size;

  char* ws = (char*)d_ws;
  // workspace layout (256B aligned): bf16 copies + bf16 intermediates
  __bf16* x_bf    = (__bf16*)(ws);                       // 4096*1280
  __bf16* qw_bf   = (__bf16*)(ws + 10485760);            // 3840*1280
  __bf16* pw_bf   = (__bf16*)(ws + 20316160);            // 1280*1280
  __bf16* qkv_bf  = (__bf16*)(ws + 23592960);            // 4096*3840
  __bf16* attn_bf = (__bf16*)(ws + 55050240);            // 4096*1280

  // 1) fp32 -> bf16 conversions
  cvt_bf16_kernel<<<2048, 256, 0, stream>>>(x, x_bf, S_ * D_);
  cvt_bf16_kernel<<<2048, 256, 0, stream>>>(qkv_w, qw_bf, DQKV * D_);
  cvt_bf16_kernel<<<2048, 256, 0, stream>>>(proj_w, pw_bf, D_ * D_);

  // 2) QKV GEMM: [4096x1280] @ [3840x1280]^T + bias -> bf16 [4096x3840]
  {
    dim3 grid(DQKV / 128, S_ / 128);
    gemm_wmma_kernel<true><<<grid, 256, 0, stream>>>(
        x_bf, qw_bf, qkv_b, qkv_bf, nullptr, S_, DQKV, D_);
  }

  // 3) RoPE on q and k, in place
  {
    const int n = S_ * H_ * (HD_ / 2);
    rope_kernel<<<(n + 255) / 256, 256, 0, stream>>>(qkv_bf, poscos, possin);
  }

  // 4) segment-local flash attention -> bf16 [4096x1280]
  {
    dim3 grid(SEG_ / 64, S_ / SEG_, H_);
    attn_kernel<<<grid, 128, 0, stream>>>(qkv_bf, attn_bf);
  }

  // 5) projection: [4096x1280] @ [1280x1280]^T + bias -> fp32 d_out
  {
    dim3 grid(D_ / 128, S_ / 128);
    gemm_wmma_kernel<false><<<grid, 256, 0, stream>>>(
        attn_bf, pw_bf, proj_b, nullptr, (float*)d_out, S_, D_, D_);
  }
}
